// State_and_Language_Pair_Encoder_25537875542358
// MI455X (gfx1250) — matrix operand order repaired and checked by reading, not compiled, not run on hardware
//
#include <hip/hip_runtime.h>
#include <hip/hip_bf16.h>
#include <math.h>

// ---------------------------------------------------------------------------
// Problem constants (from reference): B=4096, D=256, H=384, Q=512, E=34
// ---------------------------------------------------------------------------
#define B_TOT 4096
#define D_IN  256
#define H_DIM 384
#define Q_DIM 512
#define E_NUM 34
#define MTILE 32    // batch rows per block (2 x 16-row WMMA M-tiles)
#define SPAD  264   // padded LDS row stride (halves) for state tile (256+8)
#define HPAD  392   // padded LDS row stride (halves) for hidden tiles (384+8)

typedef __attribute__((ext_vector_type(16))) __bf16       v16bf;
typedef __attribute__((ext_vector_type(8)))  float        v8f;
typedef __attribute__((ext_vector_type(4)))  unsigned int v4u;

union Frag {
  v16bf bf;
  v4u   u4[2];
};

__device__ __forceinline__ unsigned short f2bf(float f) {
  unsigned int x = __builtin_bit_cast(unsigned int, f);
  unsigned int r = (x + 0x7FFFu + ((x >> 16) & 1u)) >> 16;   // RNE
  return (unsigned short)r;
}
__device__ __forceinline__ float bf2f(unsigned short h) {
  unsigned int x = ((unsigned int)h) << 16;
  return __builtin_bit_cast(float, x);
}

// ---------------------------------------------------------------------------
// Two 16x16 output tiles sharing one B fragment:
//   C0 += A0(16xK) * B(Kx16),  C1 += A1(16xK) * B(Kx16)
// A rows 0..15 -> tile 0, rows 16..31 -> tile 1 (LDS, padded stride).
// A-fragment (16-bit 16x32): lane l holds row M=l&15; halves i=0..15 map to
//   K = (i&7) + 8*(l>>4) + 16*(i>>3)  -> two contiguous 8-half (16B) runs.
// B-fragment packed so lane l's 16 halves are contiguous in memory.
// ---------------------------------------------------------------------------
template <int KTILES>
__device__ __forceinline__ void mm_tile2(const unsigned short* sIn, int inPad,
                                         const unsigned short* wt, size_t kstep,
                                         int lane, v8f& c0, v8f& c1) {
  const int ab0 = (lane & 15) * inPad + ((lane >> 4) << 3);
  const int ab1 = ab0 + 16 * inPad;
#pragma unroll
  for (int tk = 0; tk < KTILES; ++tk) {
    Frag fa0, fa1, fb;
    const unsigned short* ap0 = sIn + ab0 + tk * 32;
    const unsigned short* ap1 = sIn + ab1 + tk * 32;
    fa0.u4[0] = *(const v4u*)(ap0);          // ds_load_b128
    fa0.u4[1] = *(const v4u*)(ap0 + 16);
    fa1.u4[0] = *(const v4u*)(ap1);
    fa1.u4[1] = *(const v4u*)(ap1 + 16);
    const unsigned short* wp = wt + (size_t)tk * kstep + (size_t)lane * 16;
    fb.u4[0] = *(const v4u*)(wp);            // global_load_b128
    fb.u4[1] = *(const v4u*)(wp + 8);
    c0 = __builtin_amdgcn_wmma_f32_16x16x32_bf16(false, fa0.bf, false, fb.bf,
                                                 (short)0, c0, false, false);
    c1 = __builtin_amdgcn_wmma_f32_16x16x32_bf16(false, fa1.bf, false, fb.bf,
                                                 (short)0, c1, false, false);
  }
}

// ---------------------------------------------------------------------------
// Prep kernel 1: fp32 -> bf16 flat conversion (used for state)
// ---------------------------------------------------------------------------
__global__ void __launch_bounds__(256)
cvt_f32_to_bf16(const float* __restrict__ src, unsigned short* __restrict__ dst,
                int n) {
  const int i = blockIdx.x * 256 + threadIdx.x;
  if (i < n) dst[i] = f2bf(src[i]);
}

// ---------------------------------------------------------------------------
// Prep kernel 2: repack weights [E][K][N] fp32 into per-tile B fragments.
// Tile id = (e*KT + tk)*NT + tn ; within a tile, flat t = lane*16 + i with
//   k = tk*32 + (i&7) + 8*(lane>>4) + 16*(i>>3),  n = tn*16 + (lane&15)
// ---------------------------------------------------------------------------
__global__ void __launch_bounds__(256)
pack_wfrag(const float* __restrict__ W, unsigned short* __restrict__ dst,
           int K, int N, int KT, int NT) {
  const int tile = blockIdx.x;
  const int tn = tile % NT;
  const int tk = (tile / NT) % KT;
  const int e  = tile / (NT * KT);
  for (int t = threadIdx.x; t < 512; t += 256) {
    const int l = t >> 4;
    const int i = t & 15;
    const int k = tk * 32 + (i & 7) + ((l >> 4) << 3) + ((i >> 3) << 4);
    const int n = tn * 16 + (l & 15);
    dst[(size_t)tile * 512 + t] = f2bf(W[((size_t)e * K + k) * N + n]);
  }
}

// ---------------------------------------------------------------------------
// Fused 3-layer expert MLP, one (32-row batch tile, expert) per block.
// 8 waves; each wave owns 16-wide N-tiles; each loaded B fragment feeds
// two WMMAs (rows 0-15 and 16-31) -> 2x weight reuse vs a 16-row tile.
// LDS ping-pong: buf0 = state -> (h1 in buf1) -> (h2 back in buf0).
// ---------------------------------------------------------------------------
__global__ void __launch_bounds__(256)
fused_expert_mlp(const unsigned short* __restrict__ stateB,
                 const unsigned short* __restrict__ W1f, const float* __restrict__ b1,
                 const unsigned short* __restrict__ W2f, const float* __restrict__ b2,
                 const unsigned short* __restrict__ W3f, const float* __restrict__ b3,
                 unsigned short* __restrict__ qv_ws) {
  __shared__ unsigned short sBuf0[MTILE * HPAD];   // state, then h2
  __shared__ unsigned short sBuf1[MTILE * HPAD];   // h1

  const int e    = blockIdx.y;
  const int b0   = blockIdx.x * MTILE;
  const int tid  = threadIdx.x;
  const int lane = tid & 31;
  const int wave = tid >> 5;

  // Stage state tile: 32 rows x 256 bf16 (64B per thread).
  {
    const int r  = tid >> 3;              // 0..31
    const int c0 = (tid & 7) * 32;        // 0..224 step 32 halves
    const v4u* src = (const v4u*)(stateB + (size_t)(b0 + r) * D_IN + c0);
    v4u* dst = (v4u*)(sBuf0 + r * SPAD + c0);
    dst[0] = src[0];
    dst[1] = src[1];
    dst[2] = src[2];
    dst[3] = src[3];
  }
  __syncthreads();

  // ---- Layer 1: relu(state @ W1[e] + b1[e])  [32x384] -> sBuf1 ----
  for (int t = wave; t < 24; t += 8) {
    v8f c0 = {}, c1 = {};
    const unsigned short* wt = W1f + ((size_t)e * 8 * 24 + t) * 512;
    mm_tile2<8>(sBuf0, SPAD, wt, (size_t)24 * 512, lane, c0, c1);
    const int n = t * 16 + (lane & 15);
    const float bv = b1[e * H_DIM + n];
#pragma unroll
    for (int r = 0; r < 8; ++r) {
      const int m = r + ((lane >> 4) << 3);
      sBuf1[m * HPAD + n]        = f2bf(fmaxf(c0[r] + bv, 0.f));
      sBuf1[(16 + m) * HPAD + n] = f2bf(fmaxf(c1[r] + bv, 0.f));
    }
  }
  __syncthreads();

  // ---- Layer 2: relu(h1 @ W2[e] + b2[e])  [32x384] -> sBuf0 ----
  for (int t = wave; t < 24; t += 8) {
    v8f c0 = {}, c1 = {};
    const unsigned short* wt = W2f + ((size_t)e * 12 * 24 + t) * 512;
    mm_tile2<12>(sBuf1, HPAD, wt, (size_t)24 * 512, lane, c0, c1);
    const int n = t * 16 + (lane & 15);
    const float bv = b2[e * H_DIM + n];
#pragma unroll
    for (int r = 0; r < 8; ++r) {
      const int m = r + ((lane >> 4) << 3);
      sBuf0[m * HPAD + n]        = f2bf(fmaxf(c0[r] + bv, 0.f));
      sBuf0[(16 + m) * HPAD + n] = f2bf(fmaxf(c1[r] + bv, 0.f));
    }
  }
  __syncthreads();

  // ---- Layer 3: qv = h2 @ W3[e] + b3[e]  [32x512] -> global (bf16) ----
  for (int t = wave; t < 32; t += 8) {
    v8f c0 = {}, c1 = {};
    const unsigned short* wt = W3f + ((size_t)e * 12 * 32 + t) * 512;
    mm_tile2<12>(sBuf0, HPAD, wt, (size_t)32 * 512, lane, c0, c1);
    const int n = t * 16 + (lane & 15);
    const float bv = b3[e * Q_DIM + n];
#pragma unroll
    for (int r = 0; r < 8; ++r) {
      const int m = r + ((lane >> 4) << 3);
      qv_ws[((size_t)(b0 + m) * E_NUM + e) * Q_DIM + n]      = f2bf(c0[r] + bv);
      qv_ws[((size_t)(b0 + 16 + m) * E_NUM + e) * Q_DIM + n] = f2bf(c1[r] + bv);
    }
  }
}

// ---------------------------------------------------------------------------
// Finalize: cosine sim -> log_softmax over experts -> weighted mix + gather.
// One block per batch row; qv rows (34 x 512 bf16 = 34KB) are L2-resident.
// ---------------------------------------------------------------------------
__global__ void __launch_bounds__(256)
finalize_softmax_mix(const unsigned short* __restrict__ qv_ws,
                     const float* __restrict__ lang_table,
                     const int* __restrict__ task_id,
                     float* __restrict__ out_latent,
                     float* __restrict__ out_logp,
                     float* __restrict__ out_target) {
  const int b    = blockIdx.x;
  const int tid  = threadIdx.x;
  const int lane = tid & 31;
  const int wave = tid >> 5;
  __shared__ float sDot[E_NUM];
  __shared__ float sQn2[E_NUM];
  __shared__ float sW[E_NUM];
  __shared__ float sLogp[E_NUM];

  const int tb = task_id[b];

  // Cache the lane's slice of lang_emb[b]: q = lane*2 + 64*k, k=0..7.
  float lx[8], ly[8];
#pragma unroll
  for (int k = 0; k < 8; ++k) {
    const float* lp = lang_table + (size_t)tb * Q_DIM + lane * 2 + 64 * k;
    lx[k] = lp[0];
    ly[k] = lp[1];
  }
  // ||lang||^2 (same in every wave; xor-reduce broadcasts to all lanes).
  float ln2 = 0.f;
#pragma unroll
  for (int k = 0; k < 8; ++k) ln2 += lx[k] * lx[k] + ly[k] * ly[k];
#pragma unroll
  for (int off = 16; off > 0; off >>= 1) ln2 += __shfl_xor(ln2, off, 32);
  const float ln = sqrtf(ln2);

  // Pass 1: per-expert dot(qv, lang) and ||qv||^2. Wave w handles e=w,w+8,...
  for (int e2 = wave; e2 < E_NUM; e2 += 8) {
    const unsigned short* base = qv_ws + ((size_t)b * E_NUM + e2) * Q_DIM;
    float d = 0.f, s = 0.f;
#pragma unroll
    for (int k = 0; k < 8; ++k) {
      const unsigned int u = *(const unsigned int*)(base + lane * 2 + 64 * k);
      const float v0 = bf2f((unsigned short)(u & 0xffffu));
      const float v1 = bf2f((unsigned short)(u >> 16));
      d += v0 * lx[k] + v1 * ly[k];
      s += v0 * v0 + v1 * v1;
    }
#pragma unroll
    for (int off = 16; off > 0; off >>= 1) {
      d += __shfl_xor(d, off, 32);
      s += __shfl_xor(s, off, 32);
    }
    if (lane == 0) { sDot[e2] = d; sQn2[e2] = s; }
  }
  __syncthreads();

  // log_softmax over the 34 experts (tiny, serial on thread 0).
  if (tid == 0) {
    float sc[E_NUM];
    float mx = -1e30f;
    for (int e2 = 0; e2 < E_NUM; ++e2) {
      const float qn = sqrtf(sQn2[e2]);
      const float cs = sDot[e2] / fmaxf(qn * ln, 1e-8f);
      sc[e2] = cs * 10.0f;                 // divide by TEMP=0.1
      mx = fmaxf(mx, sc[e2]);
    }
    float sum = 0.f;
    for (int e2 = 0; e2 < E_NUM; ++e2) sum += __expf(sc[e2] - mx);
    const float lse = __logf(sum);
    for (int e2 = 0; e2 < E_NUM; ++e2) {
      const float lp = sc[e2] - mx - lse;
      sLogp[e2] = lp;
      sW[e2]    = __expf(lp);
    }
  }
  __syncthreads();
  if (tid < E_NUM) out_logp[(size_t)b * E_NUM + tid] = sLogp[tid];

  // Pass 2: latent_vec = sum_e w_e * qv[b,e,:] ; latent_target = qv[b,tb,:]
  for (int q = tid; q < Q_DIM; q += 256) {
    const unsigned short* base = qv_ws + (size_t)b * E_NUM * Q_DIM + q;
    float acc = 0.f;
#pragma unroll 2
    for (int e2 = 0; e2 < E_NUM; ++e2)
      acc += sW[e2] * bf2f(base[(size_t)e2 * Q_DIM]);
    out_latent[(size_t)b * Q_DIM + q] = acc;
    out_target[(size_t)b * Q_DIM + q] = bf2f(base[(size_t)tb * Q_DIM]);
  }
}

// ---------------------------------------------------------------------------
// Host-side launcher
// ---------------------------------------------------------------------------
extern "C" void kernel_launch(void* const* d_in, const int* in_sizes, int n_in,
                              void* d_out, int out_size, void* d_ws, size_t ws_size,
                              hipStream_t stream) {
  (void)in_sizes; (void)n_in; (void)out_size; (void)ws_size;

  const float* state = (const float*)d_in[0];
  const int*   task  = (const int*)  d_in[1];
  const float* W1    = (const float*)d_in[2];
  const float* b1    = (const float*)d_in[3];
  const float* W2    = (const float*)d_in[4];
  const float* b2    = (const float*)d_in[5];
  const float* W3    = (const float*)d_in[6];
  const float* b3    = (const float*)d_in[7];
  const float* lang  = (const float*)d_in[8];

  // Workspace carve-up (bytes, 256B aligned).
  char* ws = (char*)d_ws;
  size_t off = 0;
  auto carve = [&](size_t bytes) -> char* {
    char* p = ws + off;
    off = (off + bytes + 255) & ~(size_t)255;
    return p;
  };
  unsigned short* stateB = (unsigned short*)carve((size_t)B_TOT * D_IN * 2);
  unsigned short* W1f    = (unsigned short*)carve((size_t)E_NUM * 8  * 24 * 512 * 2);
  unsigned short* W2f    = (unsigned short*)carve((size_t)E_NUM * 12 * 24 * 512 * 2);
  unsigned short* W3f    = (unsigned short*)carve((size_t)E_NUM * 12 * 32 * 512 * 2);
  unsigned short* qv_ws  = (unsigned short*)carve((size_t)B_TOT * E_NUM * Q_DIM * 2);

  float* out_latent = (float*)d_out;
  float* out_logp   = out_latent + (size_t)B_TOT * Q_DIM;
  float* out_target = out_logp   + (size_t)B_TOT * E_NUM;

  // 1) Precision conversion + weight fragment packing.
  {
    const int n = B_TOT * D_IN;
    cvt_f32_to_bf16<<<(n + 255) / 256, 256, 0, stream>>>(state, stateB, n);
  }
  pack_wfrag<<<E_NUM * 8  * 24, 256, 0, stream>>>(W1, W1f, D_IN,  H_DIM, 8,  24);
  pack_wfrag<<<E_NUM * 12 * 24, 256, 0, stream>>>(W2, W2f, H_DIM, H_DIM, 12, 24);
  pack_wfrag<<<E_NUM * 12 * 32, 256, 0, stream>>>(W3, W3f, H_DIM, Q_DIM, 12, 32);

  // 2) Fused 3-layer WMMA MLP per (32-row batch tile, expert).
  {
    dim3 grid(B_TOT / MTILE, E_NUM);
    fused_expert_mlp<<<grid, 256, 0, stream>>>(stateB, W1f, b1, W2f, b2,
                                               W3f, b3, qv_ws);
  }

  // 3) Softmax over experts + weighted mixture + target gather.
  finalize_softmax_mix<<<B_TOT, 256, 0, stream>>>(qv_ws, lang, task,
                                                  out_latent, out_logp,
                                                  out_target);
}
